// ResizeConv2D_49503793054289
// MI455X (gfx1250) — matrix operand order repaired and checked by reading, not compiled, not run on hardware
//
#include <hip/hip_runtime.h>

// ---------------------------------------------------------------------------
// Fused 2x-upsample (transposed conv 3x3, 512->512) + [1,3,3,1] blur,
// reformulated as 4 phase-convolutions (3x3 SAME over x) and executed as an
// implicit GEMM on CDNA5 WMMA bf16 (f32 accumulate), with double-buffered
// GLOBAL_LOAD_ASYNC_TO_LDS staging (ASYNCcnt) for the A/B tiles.
// ---------------------------------------------------------------------------

typedef __attribute__((ext_vector_type(16))) __bf16 v16bf;
typedef __attribute__((ext_vector_type(8)))  __bf16 v8bf;
typedef __attribute__((ext_vector_type(8)))  float  v8f;
typedef __attribute__((ext_vector_type(4)))  unsigned int v4u;

#define IC 512
#define OC 512
#define HH 64
#define WW 64

#define AROWB 144                 // A LDS row stride: 128B data + 16B bank pad
#define BROWB 144                 // B LDS row stride
#define ABYTES (64 * AROWB)       // 9216
#define BBYTES (128 * BROWB)      // 18432
#define BUFBYTES (ABYTES + BBYTES) // 27648 per buffer, x2 buffers = 55296 B LDS

__device__ __forceinline__ __bf16 f2bf(float f) {
  union { float f; unsigned u; } in; in.f = f;
  unsigned u = in.u;
  unsigned r = (u + 0x7FFFu + ((u >> 16) & 1u)) >> 16;  // round-to-nearest-even
  union { unsigned short s; __bf16 b; } o; o.s = (unsigned short)r;
  return o.b;
}

// Aperture rule (ISA 10.2): for a generic pointer into LDS, addr[31:0] is the
// LDS byte offset -- so the async-copy LDS-dest VGPR is just the truncation.
__device__ __forceinline__ void async_copy16(const void* ldsdst, const void* gsrc) {
  unsigned l = (unsigned)(unsigned long long)ldsdst;
  asm volatile("global_load_async_to_lds_b128 %0, %1, off"
               :: "v"(l), "v"(gsrc) : "memory");
}
__device__ __forceinline__ void wait_async0() {
  asm volatile("s_wait_asynccnt 0x0" ::: "memory");
}

// x: NCHW fp32  ->  xbf: NHWC bf16  (ic contiguous == GEMM K contiguous)
__global__ __launch_bounds__(256) void prep_x_nhwc_bf16(
    const float* __restrict__ x, __bf16* __restrict__ xbf) {
  int idx = blockIdx.x * blockDim.x + threadIdx.x;   // 16,777,216 elements
  int ic = idx & (IC - 1);
  int t  = idx >> 9;
  int wc = t & (WW - 1);
  t >>= 6;
  int h = t & (HH - 1);
  int n = t >> 6;
  xbf[idx] = f2bf(x[((n * IC + ic) * HH + h) * WW + wc]);
}

// 1D mixing tables folding transposed-conv phase selection with the blur.
// APH[phase][d+1][k]: coefficient of w-tap k at x-offset d. b=[1,3,3,1]/4.
__device__ __constant__ float APH[2][3][3] = {
  {{0.75f, 0.25f, 0.00f},   // even d=-1: b1*w0 + b0*w1
   {0.25f, 0.75f, 0.75f},   // even d= 0: b3*w0 + b2*w1 + b1*w2
   {0.00f, 0.00f, 0.25f}},  // even d=+1: b3*w2
  {{0.25f, 0.00f, 0.00f},   // odd  d=-1: b0*w0
   {0.75f, 0.75f, 0.25f},   // odd  d= 0: b2*w0 + b1*w1 + b0*w2
   {0.00f, 0.25f, 0.75f}}   // odd  d=+1: b3*w1 + b2*w2
};

// w: HWIO fp32 -> wct[ph][tap][oc][ic] bf16 (K=ic contiguous per oc row)
__global__ __launch_bounds__(256) void prep_w_fused_bf16(
    const float* __restrict__ w, __bf16* __restrict__ wct) {
  int idx = blockIdx.x * blockDim.x + threadIdx.x;   // 9,437,184 elements
  int ic = idx & (IC - 1);
  int t  = idx >> 9;
  int oc = t & (OC - 1);
  t >>= 9;
  int tap = t % 9;
  int ph  = t / 9;                // 0..3
  int py = ph >> 1, px = ph & 1;
  int dy = tap / 3, dx = tap % 3; // 0..2 == d+1
  const float coef = 0.0147313913f; // 1/sqrt(3*3*512)
  float acc = 0.f;
#pragma unroll
  for (int kh = 0; kh < 3; ++kh) {
    float ay = APH[py][dy][kh];
#pragma unroll
    for (int kw = 0; kw < 3; ++kw) {
      float ax = APH[px][dx][kw];
      acc += ay * ax * w[((kh * 3 + kw) * IC + ic) * OC + oc];
    }
  }
  wct[idx] = f2bf(acc * coef);
}

// Main implicit-GEMM kernel.
// Block: 64 M (one x-row: fixed n,i; j=0..63) x 128 N (oc), one phase.
// 8 waves as 4(M) x 2(N); each wave 16M x 64N -> 4 accumulators.
// K pipeline: 72 phases (9 taps x 8 chunks of K=64), double-buffered async
// LDS staging; 8 WMMAs per wave per phase.
__global__ __launch_bounds__(256) void fused_up2_conv3_blur(
    const __bf16* __restrict__ xbf, const __bf16* __restrict__ wct,
    float* __restrict__ out) {
  __shared__ __align__(16) unsigned char lds[2 * BUFBYTES];

  const int bm = blockIdx.x;      // 0..511 -> (n, i)
  const int bn = blockIdx.y;      // 0..3   -> oc tile of 128
  const int ph = blockIdx.z;      // 0..3   -> output phase
  const int n  = bm >> 6;
  const int i  = bm & 63;
  const int py = ph >> 1, px = ph & 1;

  const int tid  = threadIdx.x;
  const int lane = tid & 31;      // wave32 lane
  const int wv   = tid >> 5;      // wave 0..7
  const int mr   = wv >> 1;       // wave M row 0..3
  const int ncw  = wv & 1;        // wave N col 0..1
  const int lsel = lane & 15;
  const int hi   = lane >> 4;

  v8f acc[4] = {};

  // Stage phase p (tap, 64-wide K chunk) into buffer bufsel via async copies.
  auto stage = [&](int p, int bufsel) {
    const int tap = p >> 3;
    const int kb  = (p & 7) * 64;                       // K base (elements)
    const int dy  = ((0x2A540 >> (tap * 2)) & 3) - 1;   // tap/3 - 1
    const int dx  = ((0x24924 >> (tap * 2)) & 3) - 1;   // tap%3 - 1
    const int yy  = i + dy;
    const bool rowv = (yy >= 0) && (yy < HH);
    unsigned char* buf = lds + bufsel * BUFBYTES;
    // A tile: 64 rows x 64 K = 8KB -> 512 x 16B chunks, 2 per thread.
#pragma unroll
    for (int s = 0; s < 2; ++s) {
      const int ch = tid + s * 256;
      const int ar = ch >> 3, acol = ch & 7;  // row j, 16B column
      const int xj = ar + dx;
      unsigned char* dst = buf + ar * AROWB + acol * 16;
      if (rowv && xj >= 0 && xj < WW) {
        const __bf16* g = xbf + (size_t)((n * HH + yy) * WW + xj) * IC + kb + acol * 8;
        async_copy16(dst, g);
      } else {
        *(v4u*)dst = (v4u){0u, 0u, 0u, 0u};   // halo zero-fill (ds path)
      }
    }
    // B tile: 128 oc rows x 64 K = 16KB -> 1024 x 16B chunks, 4 per thread.
    const __bf16* bbase = wct + (size_t)((ph * 9 + tap) * OC + bn * 128) * IC + kb;
#pragma unroll
    for (int s = 0; s < 4; ++s) {
      const int ch = tid + s * 256;
      const int br = ch >> 3, bcol = ch & 7;
      async_copy16(buf + ABYTES + br * BROWB + bcol * 16,
                   bbase + (size_t)br * IC + bcol * 8);
    }
  };

  // Batch-load all fragments first, then issue all 8 WMMAs back-to-back so
  // the scheduler can use partial s_wait_dscnt instead of full drains.
  auto compute = [&](int bufsel) {
    const unsigned char* buf = lds + bufsel * BUFBYTES;
    // A fragment (16x32 bf16): lane<16 holds K 0..7 & 16..23, lane>=16 holds
    // K 8..15 & 24..31 of row M=lsel (ISA 16-bit A layout).
    const unsigned char* apBase = buf + (mr * 16 + lsel) * AROWB + hi * 16;
    // B fragment (32x16): lane<16 = K 0..7 & 16..23, lane>=16 = K 8..15 &
    // 24..31 of column N=lsel (same K order as A).
    const unsigned char* bpBase = buf + ABYTES + (ncw * 64 + lsel) * BROWB + hi * 16;
    v16bf a[2];
    v16bf b[2][4];
#pragma unroll
    for (int ks = 0; ks < 2; ++ks) {
      const unsigned char* ap = apBase + ks * 64;
      v8bf alo = *(const v8bf*)ap;
      v8bf ahi = *(const v8bf*)(ap + 32);
      a[ks] = __builtin_shufflevector(alo, ahi,
          0, 1, 2, 3, 4, 5, 6, 7, 8, 9, 10, 11, 12, 13, 14, 15);
#pragma unroll
      for (int f = 0; f < 4; ++f) {
        const unsigned char* bp = bpBase + f * (16 * BROWB) + ks * 64;
        v8bf bl = *(const v8bf*)bp;
        v8bf bh = *(const v8bf*)(bp + 32);
        b[ks][f] = __builtin_shufflevector(bl, bh,
            0, 1, 2, 3, 4, 5, 6, 7, 8, 9, 10, 11, 12, 13, 14, 15);
      }
    }
#pragma unroll
    for (int ks = 0; ks < 2; ++ks)
#pragma unroll
      for (int f = 0; f < 4; ++f)
        acc[f] = __builtin_amdgcn_wmma_f32_16x16x32_bf16(
            false, a[ks], false, b[ks][f], (short)0, acc[f], false, false);
  };

  int cur = 0;
  stage(0, 0);
  for (int p = 0; p < 72; ++p) {
    wait_async0();                 // async fills for buf `cur` are complete
    __syncthreads();               // ...and visible to all waves
    if (p + 1 < 72) stage(p + 1, cur ^ 1);  // overlap next fill with compute
    compute(cur);
    cur ^= 1;
  }

  // D layout: lane L, VGPR q -> N = L&15, M = q + 8*(L>>4); M is j in-row.
  const int ocb = bn * 128 + ncw * 64 + lsel;
  const int jb  = mr * 16 + hi * 8;
#pragma unroll
  for (int f = 0; f < 4; ++f) {
    float* o = out + ((size_t)(n * OC + ocb + f * 16) * 128 + (2 * i + py)) * 128 + px;
#pragma unroll
    for (int q = 0; q < 8; ++q) o[2 * (jb + q)] = acc[f][q];
  }
}

extern "C" void kernel_launch(void* const* d_in, const int* in_sizes, int n_in,
                              void* d_out, int out_size, void* d_ws, size_t ws_size,
                              hipStream_t stream) {
  (void)in_sizes; (void)n_in; (void)out_size; (void)ws_size;
  const float* x = (const float*)d_in[0];   // (8,512,64,64) fp32
  const float* w = (const float*)d_in[1];   // (3,3,512,512) fp32
  float* out = (float*)d_out;               // (8,512,128,128) fp32

  // Workspace: [ xbf : 16,777,216 bf16 = 32 MiB ][ wct : 9,437,184 bf16 = 18 MiB ]
  __bf16* xbf = (__bf16*)d_ws;
  __bf16* wct = (__bf16*)((char*)d_ws + (size_t)16777216 * 2);

  prep_x_nhwc_bf16<<<16777216 / 256, 256, 0, stream>>>(x, xbf);
  prep_w_fused_bf16<<<9437184 / 256, 256, 0, stream>>>(w, wct);

  dim3 grid(512, 4, 4);  // (n*i rows, oc tiles of 128, phases)
  fused_up2_conv3_blur<<<grid, 256, 0, stream>>>(xbf, wct, out);
}
